// LaplacianKANLinear_59030030516697
// MI455X (gfx1250) — compile-verified, matrix operand order checked by reading, not run on hardware
//
#include <hip/hip_runtime.h>
#include <hip/hip_bf16.h>

typedef __attribute__((ext_vector_type(16))) _Float16 v16h;
typedef __attribute__((ext_vector_type(8)))  _Float16 v8h;
typedef __attribute__((ext_vector_type(8)))  float    v8f;

#define N_TOK  8192
#define IN_F   1024
#define OUT_F  1024
#define GRID_G 8
#define KTOT   (IN_F + IN_F * GRID_G)   // 9216
#define BM 128
#define BN 128
#define BK 32
#define NITER (KTOT / BK)               // 288

// ---------------------------------------------------------------------------
// Pack B = [ base_weight | spline_weight * spline_scaler ] into f16, shape
// [OUT_F][KTOT] row-major. One-time 9.4M-element transform into workspace.
// ---------------------------------------------------------------------------
__global__ __launch_bounds__(256) void kan_pack_b(
    const float* __restrict__ base_w,    // [OUT_F][IN_F]
    const float* __restrict__ spline_w,  // [OUT_F][IN_F][GRID_G]
    const float* __restrict__ scaler,    // [OUT_F][IN_F]
    _Float16* __restrict__ Bw)           // [OUT_F][KTOT]
{
    const int o = blockIdx.x;
    for (int k = threadIdx.x; k < KTOT; k += 256) {
        float v;
        if (k < IN_F) {
            v = base_w[(size_t)o * IN_F + k];
        } else {
            const int t = k - IN_F;
            const int i = t >> 3;
            const int g = t & 7;
            v = spline_w[((size_t)o * IN_F + i) * GRID_G + g] *
                scaler[(size_t)o * IN_F + i];
        }
        Bw[(size_t)o * KTOT + k] = (_Float16)v;
    }
}

// ---------------------------------------------------------------------------
// Fused GEMM: out[n][o] = sum_k A[n][k] * Bw[o][k]
//   A[n][k] = silu(x[n][k])                 for k < 1024
//   A[n][k] = exp(-|x[n][i] - grid[i][g]|)  for k >= 1024 (i=(k-1024)/8, g=k%8)
// Double-buffered LDS, one barrier per K-step. B tiles staged via
// GLOBAL_LOAD_ASYNC_TO_LDS_B128 (ASYNCcnt path, no VGPR round-trip);
// A tiles computed on the fly (exp/silu VALU co-executes with WMMA).
// ---------------------------------------------------------------------------
__global__ __launch_bounds__(256) void kan_gemm(
    const float* __restrict__ x,      // [N_TOK][IN_F]
    const float* __restrict__ grid,   // [IN_F][GRID_G]
    const _Float16* __restrict__ Bw,  // [OUT_F][KTOT]
    float* __restrict__ out)          // [N_TOK][OUT_F]
{
    __shared__ __align__(16) _Float16 As[2][BM * BK];   // 2 x 8 KB
    __shared__ __align__(16) _Float16 Bs[2][BN * BK];   // 2 x 8 KB

    const int tid   = threadIdx.x;
    const int lane  = tid & 31;
    const int wave  = tid >> 5;
    const int waveM = wave & 3;      // 4 waves along M
    const int waveN = wave >> 2;     // 2 waves along N
    const int m0 = blockIdx.x * BM;
    const int n0 = blockIdx.y * BN;

    // WMMA fragment addressing (16-bit A/B layout, wave32):
    //   lanes 0-15  : row = lane,    K = {0..7, 16..23}
    //   lanes 16-31 : row = lane-16, K = {8..15, 24..31}
    const int frow = lane & 15;
    const int kb   = (lane >> 4) * 8;

    // Tile-fill addressing: thread -> (row 0..127, 16-element column half)
    const int fr    = tid >> 1;
    const int fhalf = tid & 1;

    // Async B staging: one instruction offset covers both LDS and global.
    const _Float16* bsrc_row = Bw + (size_t)(n0 + fr) * KTOT + fhalf * 16;

    auto fill_b = [&](int k0, int buf) {
        unsigned lds = (unsigned)(size_t)&Bs[buf][fr * BK + fhalf * 16];
        const _Float16* src = bsrc_row + k0;
        asm volatile(
            "global_load_async_to_lds_b128 %0, %1, off\n\t"
            "global_load_async_to_lds_b128 %0, %1, off offset:16"
            :: "v"(lds), "v"(src) : "memory");
    };

    auto fill_a = [&](int k0, int buf) {
        v8h h0, h1;
        if (k0 < IN_F) {
            // base path: silu(x)
            const float* xp = x + (size_t)(m0 + fr) * IN_F + k0 + fhalf * 16;
#pragma unroll
            for (int j = 0; j < 8; ++j) {
                float v = xp[j];
                h0[j] = (_Float16)(v / (1.0f + __expf(-v)));
            }
#pragma unroll
            for (int j = 0; j < 8; ++j) {
                float v = xp[8 + j];
                h1[j] = (_Float16)(v / (1.0f + __expf(-v)));
            }
        } else {
            // spline path: Laplacian kernel. 32 K-columns = 4 input features.
            const int i = ((k0 - IN_F) >> 3) + fhalf * 2;   // this thread: i, i+1
            const float xv0 = x[(size_t)(m0 + fr) * IN_F + i];
            const float xv1 = x[(size_t)(m0 + fr) * IN_F + i + 1];
            const float* gp = grid + (size_t)i * GRID_G;
#pragma unroll
            for (int g = 0; g < 8; ++g)
                h0[g] = (_Float16)__expf(-fabsf(xv0 - gp[g]));
#pragma unroll
            for (int g = 0; g < 8; ++g)
                h1[g] = (_Float16)__expf(-fabsf(xv1 - gp[GRID_G + g]));
        }
        *(v8h*)&As[buf][fr * BK + fhalf * 16]     = h0;
        *(v8h*)&As[buf][fr * BK + fhalf * 16 + 8] = h1;
    };

    v8f acc[2][4];
#pragma unroll
    for (int mi = 0; mi < 2; ++mi)
#pragma unroll
        for (int ni = 0; ni < 4; ++ni)
            acc[mi][ni] = (v8f){};

    // ---- prologue: stage 0 ----
    fill_b(0, 0);
    fill_a(0, 0);
    asm volatile("s_wait_asynccnt 0x0" ::: "memory");
    __syncthreads();

    for (int it = 0; it < NITER; ++it) {
        const int buf = it & 1;

        // ---- consume fragments of current stage ----
        v16h afr[2], bfr[4];
#pragma unroll
        for (int mi = 0; mi < 2; ++mi) {
            const int r = waveM * 32 + mi * 16 + frow;
            v8h lo = *(const v8h*)&As[buf][r * BK + kb];
            v8h hi = *(const v8h*)&As[buf][r * BK + kb + 16];
            afr[mi] = __builtin_shufflevector(lo, hi,
                0, 1, 2, 3, 4, 5, 6, 7, 8, 9, 10, 11, 12, 13, 14, 15);
        }
#pragma unroll
        for (int ni = 0; ni < 4; ++ni) {
            const int r = waveN * 64 + ni * 16 + frow;
            v8h lo = *(const v8h*)&Bs[buf][r * BK + kb];
            v8h hi = *(const v8h*)&Bs[buf][r * BK + kb + 16];
            bfr[ni] = __builtin_shufflevector(lo, hi,
                0, 1, 2, 3, 4, 5, 6, 7, 8, 9, 10, 11, 12, 13, 14, 15);
        }

        // ---- prefetch next stage into the other buffer (overlaps WMMA) ----
        if (it + 1 < NITER) {
            fill_b((it + 1) * BK, buf ^ 1);
            fill_a((it + 1) * BK, buf ^ 1);
        }

        // ---- matrix math on the XDL pipe ----
#pragma unroll
        for (int mi = 0; mi < 2; ++mi)
#pragma unroll
            for (int ni = 0; ni < 4; ++ni)
                acc[mi][ni] = __builtin_amdgcn_wmma_f32_16x16x32_f16(
                    false, afr[mi], false, bfr[ni],
                    (short)0, acc[mi][ni], false, false);

        // own async B-writes must land before the group-wide barrier
        asm volatile("s_wait_asynccnt 0x0" ::: "memory");
        __syncthreads();
    }

    // ---- epilogue: C/D layout -> global f32 ----
    const int chalf = lane >> 4;     // 0: M rows 0-7, 1: M rows 8-15
    const int ccol  = lane & 15;
#pragma unroll
    for (int mi = 0; mi < 2; ++mi)
#pragma unroll
        for (int ni = 0; ni < 4; ++ni)
#pragma unroll
            for (int v = 0; v < 8; ++v) {
                const int row = m0 + waveM * 32 + mi * 16 + v + chalf * 8;
                const int col = n0 + waveN * 64 + ni * 16 + ccol;
                out[(size_t)row * OUT_F + col] = acc[mi][ni][v];
            }
}

// ---------------------------------------------------------------------------
extern "C" void kernel_launch(void* const* d_in, const int* in_sizes, int n_in,
                              void* d_out, int out_size, void* d_ws, size_t ws_size,
                              hipStream_t stream) {
    (void)in_sizes; (void)n_in; (void)out_size; (void)ws_size;
    const float* x        = (const float*)d_in[0];
    const float* base_w   = (const float*)d_in[1];
    const float* spline_w = (const float*)d_in[2];
    const float* scaler   = (const float*)d_in[3];
    const float* grid     = (const float*)d_in[4];
    float* out            = (float*)d_out;
    _Float16* Bw          = (_Float16*)d_ws;   // OUT_F * KTOT * 2B = 18 MB

    kan_pack_b<<<OUT_F, 256, 0, stream>>>(base_w, spline_w, scaler, Bw);

    dim3 gdim(N_TOK / BM, OUT_F / BN);   // 64 x 8 workgroups
    kan_gemm<<<gdim, 256, 0, stream>>>(x, grid, Bw, out);
}
